// SpectralWiseAttention_34857954574946
// MI455X (gfx1250) — matrix operand order repaired and checked by reading, not compiled, not run on hardware
//
#include <hip/hip_runtime.h>
#include <hip/hip_bf16.h>

typedef __bf16 bf16_t;
typedef bf16_t v16bf __attribute__((ext_vector_type(16)));
typedef float  v8f   __attribute__((ext_vector_type(8)));

#define BB 4
#define CC 192
#define HW 16384
#define C3 576
#define NCHUNK 16
#define CHLEN (HW / NCHUNK)
#define EPSN 1e-12f

__device__ __forceinline__ v16bf make_frag(uint4 lo, uint4 hi) {
  union { uint4 u[2]; v16bf v; } f;
  f.u[0] = lo; f.u[1] = hi;
  return f.v;
}

__device__ __forceinline__ uint4 ldg4(const bf16_t* p) {
  return *reinterpret_cast<const uint4*>(p);
}

__device__ __forceinline__ unsigned pack_bf16x2(float a, float b) {
  union { bf16_t h[2]; unsigned u; } pk;
  pk.h[0] = (bf16_t)a;  // hardware cvt, RTNE
  pk.h[1] = (bf16_t)b;
  return pk.u;
}

// ---------------- init: zero norm2 + S accumulators ----------------
__global__ void k_init(float* __restrict__ norm2, float* __restrict__ S) {
  int i = blockIdx.x * blockDim.x + threadIdx.x;
  if (i < BB * 2 * CC) norm2[i] = 0.f;
  for (int j = i; j < BB * CC * CC; j += gridDim.x * blockDim.x) S[j] = 0.f;
}

// ---------------- convert weights to transposed bf16 ----------------
__global__ void k_convw(const float* __restrict__ Wqkv, const float* __restrict__ Wlin,
                        bf16_t* __restrict__ WqkvT, bf16_t* __restrict__ WlinT) {
  int i = blockIdx.x * blockDim.x + threadIdx.x;
  if (i < CC * C3) {                     // WqkvT[j][c] = Wqkv[c][j]
    int j = i / CC, c = i % CC;
    WqkvT[i] = (bf16_t)Wqkv[c * C3 + j];
  }
  int k = i - CC * C3;
  if (k >= 0 && k < CC * CC) {           // WlinT[e][d] = Wlin[d][e]
    int e = k / CC, d = k % CC;
    WlinT[k] = (bf16_t)Wlin[d * CC + e];
  }
}

// ---------------- QKV GEMM: qkv[n,j] = sum_c x[b,c,n] * Wqkv[c,j] ----------------
// x tile staged fp32 via async copy engine (ASYNCcnt), then packed-converted
// to a transposed bf16 A-tile. Writes qT/kT as [b][c][n], v as [b][n][c].
__global__ __launch_bounds__(256) void k_qkv(
    const float* __restrict__ x, const bf16_t* __restrict__ WqkvT,
    bf16_t* __restrict__ qT, bf16_t* __restrict__ kT,
    bf16_t* __restrict__ vbf, float* __restrict__ norm2) {
  __shared__ float  lds_x[32 * 128];   // raw [cc][nn] fp32 tile (async dest)
  __shared__ bf16_t lds_a[128 * 32];   // transposed [nn][cc] bf16 A-tile
  const int b = blockIdx.z;
  const int n0 = blockIdx.x * 128;
  const int j0 = blockIdx.y * 64;
  const int tid = threadIdx.x;
  const int lane = tid & 31, wave = tid >> 5;
  const int wm = wave & 3, wn = wave >> 2;
  const int half = lane >> 4, l16 = lane & 15;

  v8f acc[2][2] = {};
  for (int kk0 = 0; kk0 < CC; kk0 += 32) {
    __syncthreads();
    // 1) async-copy the fp32 tile straight into LDS (no VGPR roundtrip)
    for (int i = 0; i < 4; ++i) {
      int idx = i * 256 + tid;               // 0..1023 b128 transfers
      int cc = idx >> 5;                     // 0..31
      int n4 = idx & 31;                     // 0..31 (x4 floats)
      const float* gp = x + (size_t)(b * CC + kk0 + cc) * HW + n0 + n4 * 4;
      unsigned loff = (unsigned)(uintptr_t)(&lds_x[cc * 128 + n4 * 4]);
      asm volatile("global_load_async_to_lds_b128 %0, %1, off"
                   :: "v"(loff), "v"(gp) : "memory");
    }
    asm volatile("s_wait_asynccnt 0x0" ::: "memory");
    __syncthreads();
    // 2) transpose + packed bf16 convert: one b32 store per two elements
    for (int i = 0; i < 8; ++i) {
      int idx = i * 256 + tid;               // 0..2047 pairs
      int cp = idx & 15;                     // cc pair index
      int nn = idx >> 4;                     // 0..127
      float f0 = lds_x[(cp * 2) * 128 + nn];
      float f1 = lds_x[(cp * 2 + 1) * 128 + nn];
      reinterpret_cast<unsigned*>(lds_a)[nn * 16 + cp] = pack_bf16x2(f0, f1);
    }
    __syncthreads();
    // 3) fragments + WMMA
    v16bf afr[2], bfr[2];
    for (int sm = 0; sm < 2; ++sm) {
      const bf16_t* ap = &lds_a[(wm * 32 + sm * 16 + l16) * 32 + half * 8];
      afr[sm] = make_frag(*(const uint4*)ap, *(const uint4*)(ap + 16));
    }
    for (int sn = 0; sn < 2; ++sn) {
      int j = j0 + wn * 32 + sn * 16 + l16;
      const bf16_t* bp = WqkvT + (size_t)j * CC + kk0 + half * 8;
      bfr[sn] = make_frag(ldg4(bp), ldg4(bp + 16));
    }
    for (int sm = 0; sm < 2; ++sm)
      for (int sn = 0; sn < 2; ++sn)
        acc[sm][sn] = __builtin_amdgcn_wmma_f32_16x16x32_bf16(
            false, afr[sm], false, bfr[sn], (short)0, acc[sm][sn], false, false);
  }
  for (int sm = 0; sm < 2; ++sm)
    for (int sn = 0; sn < 2; ++sn) {
      int j = j0 + wn * 32 + sn * 16 + l16;
      float ss = 0.f;
      for (int r = 0; r < 8; ++r) {
        int n = n0 + wm * 32 + sm * 16 + r + half * 8;
        float val = acc[sm][sn][r];
        if (j < CC) {
          qT[(size_t)(b * CC + j) * HW + n] = (bf16_t)val;
          ss += val * val;
        } else if (j < 2 * CC) {
          kT[(size_t)(b * CC + (j - CC)) * HW + n] = (bf16_t)val;
          ss += val * val;
        } else {
          vbf[((size_t)(b * HW) + n) * CC + (j - 2 * CC)] = (bf16_t)val;
        }
      }
      if (j < 2 * CC) atomicAdd(&norm2[b * 2 * CC + j], ss);
    }
}

// ---------------- S[b,ck,cq] = sum_n k[n,ck] * q[n,cq] (split-K + f32 atomics) ----------------
__global__ __launch_bounds__(256) void k_kq(
    const bf16_t* __restrict__ kT, const bf16_t* __restrict__ qT,
    float* __restrict__ S) {
  const int lane = threadIdx.x & 31;
  const int gw = blockIdx.x * 8 + (threadIdx.x >> 5);
  const int chunk = gw & (NCHUNK - 1);
  const int tile = (gw >> 4) % 144;
  const int b = gw / (NCHUNK * 144);
  const int tm = tile / 12, tn = tile % 12;
  const int half = lane >> 4, l16 = lane & 15;
  const bf16_t* ka = kT + (size_t)(b * CC + tm * 16 + l16) * HW + chunk * CHLEN;
  const bf16_t* qb = qT + (size_t)(b * CC + tn * 16 + l16) * HW + chunk * CHLEN;
  v8f acc = {};
  for (int s = 0; s < CHLEN / 32; ++s) {
    const bf16_t* ap = ka + s * 32 + half * 8;
    const bf16_t* bp = qb + s * 32 + half * 8;
    __builtin_prefetch(ap + 256, 0, 1);   // -> global_prefetch_b8
    __builtin_prefetch(bp + 256, 0, 1);
    v16bf af = make_frag(ldg4(ap), ldg4(ap + 16));
    v16bf bq = make_frag(ldg4(bp), ldg4(bp + 16));
    acc = __builtin_amdgcn_wmma_f32_16x16x32_bf16(false, af, false, bq, (short)0, acc, false, false);
  }
  for (int r = 0; r < 8; ++r) {
    int ck = tm * 16 + r + half * 8;
    int cq = tn * 16 + l16;
    atomicAdd(&S[((size_t)b * CC + ck) * CC + cq], acc[r]);
  }
}

// ---------------- fused scale (sigma / norms) + row softmax; store attn^T bf16 ----------------
__global__ __launch_bounds__(256) void k_softmax(
    const float* __restrict__ S, const float* __restrict__ norm2,
    const float* __restrict__ sigma, bf16_t* __restrict__ attnT) {
  __shared__ float red[256];
  const int c = blockIdx.x;   // k-channel row
  const int b = blockIdx.y;
  const int d = threadIdx.x;
  const float sg = sigma[0];
  float val = -1e30f;
  if (d < CC) {
    float nk = fmaxf(sqrtf(norm2[b * 2 * CC + CC + c]), EPSN);
    float nq = fmaxf(sqrtf(norm2[b * 2 * CC + d]), EPSN);
    val = S[((size_t)b * CC + c) * CC + d] * sg / (nk * nq);
  }
  red[d] = val; __syncthreads();
  for (int s = 128; s > 0; s >>= 1) { if (d < s) red[d] = fmaxf(red[d], red[d + s]); __syncthreads(); }
  float m = red[0]; __syncthreads();
  float p = (d < CC) ? __expf(val - m) : 0.f;
  red[d] = p; __syncthreads();
  for (int s = 128; s > 0; s >>= 1) { if (d < s) red[d] += red[d + s]; __syncthreads(); }
  float inv = 1.f / red[0];
  if (d < CC) attnT[((size_t)b * CC + d) * CC + c] = (bf16_t)(p * inv);  // store transposed
}

// ---------------- fused (V @ attn) @ Wlin with LDS intermediate; write [b,c,hw] ----------------
__global__ __launch_bounds__(256) void k_out(
    const bf16_t* __restrict__ vbf, const bf16_t* __restrict__ attnT,
    const bf16_t* __restrict__ WlinT, float* __restrict__ out) {
  __shared__ bf16_t mid[128 * 192];
  const int b = blockIdx.y;
  const int n0 = blockIdx.x * 128;
  const int lane = threadIdx.x & 31, wave = threadIdx.x >> 5;
  const int half = lane >> 4, l16 = lane & 15;
  v8f acc[12];
  for (int t = 0; t < 12; ++t) acc[t] = v8f{};
  const bf16_t* arow = vbf + ((size_t)(b * HW) + n0 + wave * 16 + l16) * CC;
  for (int ks = 0; ks < 6; ++ks) {
    int kk0 = ks * 32;
    const bf16_t* ap = arow + kk0 + half * 8;
    v16bf af = make_frag(ldg4(ap), ldg4(ap + 16));
    for (int t = 0; t < 12; ++t) {
      const bf16_t* bp = attnT + ((size_t)b * CC + t * 16 + l16) * CC + kk0 + half * 8;
      v16bf bq = make_frag(ldg4(bp), ldg4(bp + 16));
      acc[t] = __builtin_amdgcn_wmma_f32_16x16x32_bf16(false, af, false, bq, (short)0, acc[t], false, false);
    }
  }
  for (int t = 0; t < 12; ++t)
    for (int r = 0; r < 8; ++r)
      mid[(wave * 16 + r + half * 8) * 192 + t * 16 + l16] = (bf16_t)acc[t][r];
  __syncthreads();
  for (int t = 0; t < 12; ++t) acc[t] = v8f{};
  const bf16_t* mrow = &mid[(wave * 16 + l16) * 192];
  for (int ks = 0; ks < 6; ++ks) {
    int kk0 = ks * 32;
    const bf16_t* ap = mrow + kk0 + half * 8;
    v16bf af = make_frag(*(const uint4*)ap, *(const uint4*)(ap + 16));
    for (int t = 0; t < 12; ++t) {
      const bf16_t* bp = WlinT + (size_t)(t * 16 + l16) * CC + kk0 + half * 8;
      v16bf bq = make_frag(ldg4(bp), ldg4(bp + 16));
      acc[t] = __builtin_amdgcn_wmma_f32_16x16x32_bf16(false, af, false, bq, (short)0, acc[t], false, false);
    }
  }
  for (int t = 0; t < 12; ++t)
    for (int r = 0; r < 8; ++r) {
      int e = t * 16 + l16;
      int n = n0 + wave * 16 + r + half * 8;
      out[((size_t)(b * CC) + e) * HW + n] = acc[t][r];
    }
}

extern "C" void kernel_launch(void* const* d_in, const int* in_sizes, int n_in,
                              void* d_out, int out_size, void* d_ws, size_t ws_size,
                              hipStream_t stream) {
  (void)in_sizes; (void)n_in; (void)out_size; (void)ws_size;
  const float* x     = (const float*)d_in[0];
  const float* Wqkv  = (const float*)d_in[1];
  const float* Wlin  = (const float*)d_in[2];
  const float* sigma = (const float*)d_in[3];
  float* out = (float*)d_out;

  char* ws = (char*)d_ws;
  size_t off = 0;
  auto take = [&](size_t bytes) -> char* {
    char* p = ws + off;
    off += (bytes + 255) & ~(size_t)255;
    return p;
  };
  bf16_t* qT    = (bf16_t*)take((size_t)BB * CC * HW * 2);
  bf16_t* kT    = (bf16_t*)take((size_t)BB * CC * HW * 2);
  bf16_t* vbf   = (bf16_t*)take((size_t)BB * HW * CC * 2);
  bf16_t* WqkvT = (bf16_t*)take((size_t)C3 * CC * 2);
  bf16_t* WlinT = (bf16_t*)take((size_t)CC * CC * 2);
  float*  norm2 = (float*)take((size_t)BB * 2 * CC * 4);
  float*  S     = (float*)take((size_t)BB * CC * CC * 4);
  bf16_t* attnT = (bf16_t*)take((size_t)BB * CC * CC * 2);

  k_init<<<576, 256, 0, stream>>>(norm2, S);
  k_convw<<<576, 256, 0, stream>>>(Wqkv, Wlin, WqkvT, WlinT);
  k_qkv<<<dim3(HW / 128, C3 / 64, BB), 256, 0, stream>>>(x, WqkvT, qT, kT, vbf, norm2);
  k_kq<<<(BB * 144 * NCHUNK) / 8, 256, 0, stream>>>(kT, qT, S);
  k_softmax<<<dim3(CC, BB), 256, 0, stream>>>(S, norm2, sigma, attnT);
  k_out<<<dim3(HW / 128, BB), 256, 0, stream>>>(vbf, attnT, WlinT, out);
}
